// UE_GAT_UD_54520314856231
// MI455X (gfx1250) — matrix operand order repaired and checked by reading, not compiled, not run on hardware
//
#include <hip/hip_runtime.h>
#include <hip/hip_bf16.h>
#include <math.h>

// ---------------------------------------------------------------------------
// GATv2 2-layer forward for MI455X (gfx1250, wave32, WMMA).
//   * Projection GEMMs in bf16 via v_wmma_f32_16x16x32_bf16 (f32 accum),
//     double-buffered LDS fed by async global->LDS DMA (ASYNCcnt).
//   * Edge phase (scores, segment softmax, aggregation) in f32 with
//     single-instruction HW atomics (global_atomic_add_f32 / max_i32).
//   * W_src/W_dst fused into one GEMM (concatenated output columns).
// ---------------------------------------------------------------------------

#define N_NODES 30000
#define N_EDGES 120000
#define D_IN    384
#define H1      8
#define D1      384
#define HD1     (H1 * D1)      // 3072
#define D2      384
#define NEG_SLOPE 0.2f

typedef __bf16 bf16_t;
typedef bf16_t v16bf __attribute__((ext_vector_type(16)));
typedef bf16_t v8bf  __attribute__((ext_vector_type(8)));
typedef float  v8f   __attribute__((ext_vector_type(8)));

__device__ __forceinline__ void wait_async0() {
#if __has_builtin(__builtin_amdgcn_s_wait_asynccnt)
    __builtin_amdgcn_s_wait_asynccnt(0);
#else
    asm volatile("s_wait_asynccnt 0x0" ::: "memory");
#endif
}

// async copy of 32 contiguous bytes global -> LDS (per lane), ASYNCcnt-tracked
__device__ __forceinline__ void async_copy32(unsigned lds_addr, const void* gptr) {
    unsigned long long ga = (unsigned long long)(uintptr_t)gptr;
    asm volatile(
        "global_load_async_to_lds_b128 %0, %1, off\n\t"
        "global_load_async_to_lds_b128 %0, %1, off offset:16"
        :: "v"(lds_addr), "v"(ga) : "memory");
}

__device__ __forceinline__ float atomAddF32(float* p, float v) {
    return __hip_atomic_fetch_add(p, v, __ATOMIC_RELAXED, __HIP_MEMORY_SCOPE_AGENT);
}

// ---------------------------------------------------------------------------
// Utility kernels
// ---------------------------------------------------------------------------
__global__ void fill_f32(float* __restrict__ p, float v, size_t n) {
    size_t i = (size_t)blockIdx.x * blockDim.x + threadIdx.x;
    if (i < n) p[i] = v;
}

__global__ void f32_to_bf16(const float* __restrict__ x, bf16_t* __restrict__ y, size_t n) {
    size_t i = (size_t)blockIdx.x * blockDim.x + threadIdx.x;
    if (i < n) y[i] = (bf16_t)x[i];
}

// W [K][Nout] f32 row-major  ->  Wt [Nout][K] bf16 row-major (transposed)
__global__ __launch_bounds__(256)
void convert_w_transpose(const float* __restrict__ W, bf16_t* __restrict__ Wt,
                         int K, int Nout) {
    __shared__ float t[32][33];
    const int tx = threadIdx.x & 31;
    const int ty = threadIdx.x >> 5;           // 0..7
    const int n = blockIdx.x * 32 + tx;
    for (int i = ty; i < 32; i += 8) {
        int k = blockIdx.y * 32 + i;
        t[i][tx] = (k < K && n < Nout) ? W[(size_t)k * Nout + n] : 0.f;
    }
    __syncthreads();
    for (int i = ty; i < 32; i += 8) {
        int wn = blockIdx.x * 32 + i;          // Wt row (output col of W)
        int wk = blockIdx.y * 32 + tx;         // Wt col (K)
        if (wn < Nout && wk < K)
            Wt[(size_t)wn * K + wk] = (bf16_t)t[tx][i];
    }
}

// ---------------------------------------------------------------------------
// bf16 WMMA GEMM:  C[M][N] (f32) = A[M][K] (bf16, row-major) * Bt[N][K]^T
// 256 threads = 8 waves; block tile 128x128; K-step 32.
// Wave tile 32x64 = 2x4 WMMA accumulators. Double-buffered LDS, async DMA.
// Requires: K % 32 == 0, N % 128 == 0, M >= 1.
// ---------------------------------------------------------------------------
#define BM 128
#define BN 128
#define BK 32
#define LDP (BK + 8)   // padded LDS row (bf16 units) -> 80B row stride

__global__ __launch_bounds__(256)
void gemm_bf16_wmma(const bf16_t* __restrict__ A, const bf16_t* __restrict__ Bt,
                    float* __restrict__ C, int M, int N, int K) {
    __shared__ bf16_t sA[2][BM][LDP];
    __shared__ bf16_t sB[2][BN][LDP];

    const int tid  = threadIdx.x;
    const int m0   = blockIdx.y * BM;
    const int n0   = blockIdx.x * BN;
    const int wave = tid >> 5;
    const int lane = tid & 31;
    const int l    = lane & 15;
    const int hi   = lane >> 4;                // half-wave select
    const int wm   = wave & 3;                 // 4 waves along M
    const int wn   = wave >> 2;                // 2 waves along N

    v8f acc[2][4];
#pragma unroll
    for (int i = 0; i < 2; i++)
#pragma unroll
        for (int j = 0; j < 4; j++)
            acc[i][j] = (v8f){0.f, 0.f, 0.f, 0.f, 0.f, 0.f, 0.f, 0.f};

    // staging: each thread DMAs 32B of A and 32B of B per K-tile.
    // Branch-free: row index clamped (duplicate rows for M-overhang; the
    // epilogue guards stores, so the extra rows are never written back).
    const int srow = tid >> 1;                 // 0..127
    const int scol = (tid & 1) * 16;           // 0 or 16 (bf16 units)
    const bf16_t* gA = A + (size_t)min(m0 + srow, M - 1) * K + scol;
    const bf16_t* gB = Bt + (size_t)min(n0 + srow, N - 1) * K + scol;
    const unsigned laA[2] = { (unsigned)(uintptr_t)&sA[0][srow][scol],
                              (unsigned)(uintptr_t)&sA[1][srow][scol] };
    const unsigned laB[2] = { (unsigned)(uintptr_t)&sB[0][srow][scol],
                              (unsigned)(uintptr_t)&sB[1][srow][scol] };

    const int ntiles = K / BK;

    // prologue: stage tile 0 into buffer 0
    async_copy32(laA[0], gA);
    async_copy32(laB[0], gB);
    wait_async0();
    __syncthreads();

    int buf = 0;
    for (int kt = 0; kt < ntiles; ++kt) {
        // issue DMA for next tile into the other buffer (overlaps with WMMA)
        if (kt + 1 < ntiles) {
            async_copy32(laA[buf ^ 1], gA + (kt + 1) * BK);
            async_copy32(laB[buf ^ 1], gB + (kt + 1) * BK);
        }

        // ---- fragments ----
        // A 16x32 bf16: lanes 0-15 row M=l hold K 0..7 | 16..23;
        // lanes 16-31 hold K 8..15 | 24..31  -> [hi*8 .. +7], [16+hi*8 .. +7]
        v16bf af[2];
#pragma unroll
        for (int mi = 0; mi < 2; mi++) {
            const int r = wm * 32 + mi * 16 + l;
            union { v16bf v; v8bf h[2]; } u;
            u.h[0] = *(const v8bf*)&sA[buf][r][hi * 8];
            u.h[1] = *(const v8bf*)&sA[buf][r][16 + hi * 8];
            af[mi] = u.v;
        }
        // B 32x16 bf16 (col-major, staged as Bt rows): lanes 0-15 col N=l hold
        // K 0..7 | 16..23; lanes 16-31 hold K 8..15 | 24..31  -> [hi*8 .. +7], [16+hi*8 .. +7]
        v16bf bf[4];
#pragma unroll
        for (int nj = 0; nj < 4; nj++) {
            const int c = wn * 64 + nj * 16 + l;
            union { v16bf v; v8bf h[2]; } u;
            u.h[0] = *(const v8bf*)&sB[buf][c][hi * 8];
            u.h[1] = *(const v8bf*)&sB[buf][c][16 + hi * 8];
            bf[nj] = u.v;
        }

        // ---- 8 WMMAs per wave per K step ----
#pragma unroll
        for (int mi = 0; mi < 2; mi++)
#pragma unroll
            for (int nj = 0; nj < 4; nj++)
                acc[mi][nj] = __builtin_amdgcn_wmma_f32_16x16x32_bf16(
                    false, af[mi], false, bf[nj], (short)0, acc[mi][nj],
                    false, false);

        wait_async0();       // own DMAs for next buffer complete
        __syncthreads();     // whole workgroup done writing next / reading cur
        buf ^= 1;
    }

    // ---- epilogue: D layout: VGPR r -> lanes0-15 M=r, lanes16-31 M=8+r; N=l
#pragma unroll
    for (int mi = 0; mi < 2; mi++) {
        const int rbase = m0 + wm * 32 + mi * 16 + hi * 8;
#pragma unroll
        for (int nj = 0; nj < 4; nj++) {
            const int col = n0 + wn * 64 + nj * 16 + l;
#pragma unroll
            for (int r = 0; r < 8; r++) {
                const int row = rbase + r;
                if (row < M)
                    C[(size_t)row * N + col] = acc[mi][nj][r];
            }
        }
    }
}

// ---------------------------------------------------------------------------
// Edge-phase kernels (f32). Order-preserving float atomic max via int atomics.
// ---------------------------------------------------------------------------
__device__ __forceinline__ void atomicMaxFloat(float* addr, float val) {
    if (!(val < 0.f))
        atomicMax((int*)addr, __float_as_int(val));
    else
        atomicMin((unsigned int*)addr, __float_as_uint(val));
}

// Layer-1 score: block = 1 edge, 8 waves = 8 heads, 32 lanes over D=384.
__global__ __launch_bounds__(256)
void edge_score_h8(const float* __restrict__ F1,   // [N][2*HD1]: fs | fd
                   const float* __restrict__ a1,   // [H1][D1]
                   const int* __restrict__ src, const int* __restrict__ dst,
                   float* __restrict__ score,      // [E][H1]
                   float* __restrict__ smax,       // [N][H1]
                   int E) {
    const int e = blockIdx.x;
    if (e >= E) return;
    const int h = threadIdx.x >> 5;
    const int lane = threadIdx.x & 31;
    const int s = src[e], d = dst[e];
    const float* fs = F1 + (size_t)s * (2 * HD1) + h * D1;
    const float* fd = F1 + (size_t)d * (2 * HD1) + HD1 + h * D1;
    const float* ah = a1 + h * D1;
    float acc = 0.f;
#pragma unroll 3
    for (int i = lane; i < D1; i += 32) {
        float v = fs[i] + fd[i];
        v = v > 0.f ? v : NEG_SLOPE * v;
        acc += v * ah[i];
    }
#pragma unroll
    for (int off = 16; off > 0; off >>= 1) acc += __shfl_xor(acc, off, 32);
    if (lane == 0) {
        score[(size_t)e * H1 + h] = acc;
        atomicMaxFloat(&smax[(size_t)d * H1 + h], acc);
    }
}

// Layer-2 score: block = 8 edges (1 wave each), H=1, D=384.
__global__ __launch_bounds__(256)
void edge_score_h1(const float* __restrict__ F2,   // [N][2*D2]: fs | fd
                   const float* __restrict__ a2,   // [D2]
                   const int* __restrict__ src, const int* __restrict__ dst,
                   float* __restrict__ score, float* __restrict__ smax, int E) {
    const int e = blockIdx.x * 8 + (threadIdx.x >> 5);
    if (e >= E) return;
    const int lane = threadIdx.x & 31;
    const int s = src[e], d = dst[e];
    const float* fs = F2 + (size_t)s * (2 * D2);
    const float* fd = F2 + (size_t)d * (2 * D2) + D2;
    float acc = 0.f;
#pragma unroll 3
    for (int i = lane; i < D2; i += 32) {
        float v = fs[i] + fd[i];
        v = v > 0.f ? v : NEG_SLOPE * v;
        acc += v * a2[i];
    }
#pragma unroll
    for (int off = 16; off > 0; off >>= 1) acc += __shfl_xor(acc, off, 32);
    if (lane == 0) {
        score[e] = acc;
        atomicMaxFloat(&smax[d], acc);
    }
}

template <int H>
__global__ void edge_exp(const float* __restrict__ score,
                         const int* __restrict__ dst,
                         const float* __restrict__ smax,
                         float* __restrict__ ex, float* __restrict__ denom, int E) {
    const int i = blockIdx.x * blockDim.x + threadIdx.x;
    if (i >= E * H) return;
    const int e = i / H, h = i - e * H;
    const float v = expf(score[i] - smax[(size_t)dst[e] * H + h]);
    ex[i] = v;
    atomAddF32(&denom[(size_t)dst[e] * H + h], v);
}

// Layer-1 aggregation: block = 1 edge, 256 threads over 3072 features.
__global__ __launch_bounds__(256)
void edge_agg_h8(const float* __restrict__ F1, const float* __restrict__ ex,
                 const float* __restrict__ denom,
                 const int* __restrict__ src, const int* __restrict__ dst,
                 float* __restrict__ agg, int E) {
    const int e = blockIdx.x;
    if (e >= E) return;
    const int s = src[e], d = dst[e];
    __shared__ float alpha[H1];
    if (threadIdx.x < H1)
        alpha[threadIdx.x] = ex[(size_t)e * H1 + threadIdx.x] /
                             denom[(size_t)d * H1 + threadIdx.x];
    __syncthreads();
    const float* fs = F1 + (size_t)s * (2 * HD1);
    float* ad = agg + (size_t)d * HD1;
#pragma unroll
    for (int i = threadIdx.x; i < HD1; i += 256)
        atomAddF32(&ad[i], alpha[i / D1] * fs[i]);
}

// Layer-2 aggregation into d_out directly (H=1).
__global__ __launch_bounds__(256)
void edge_agg_h1(const float* __restrict__ F2, const float* __restrict__ ex,
                 const float* __restrict__ denom,
                 const int* __restrict__ src, const int* __restrict__ dst,
                 float* __restrict__ out, int E) {
    const int e = blockIdx.x;
    if (e >= E) return;
    const int s = src[e], d = dst[e];
    const float alpha = ex[e] / denom[d];
    const float* fs = F2 + (size_t)s * (2 * D2);
    float* od = out + (size_t)d * D2;
#pragma unroll
    for (int i = threadIdx.x; i < D2; i += 256)
        atomAddF32(&od[i], alpha * fs[i]);
}

// g = elu(agg + b1) converted to bf16 (input activation for layer 2)
__global__ void elu_bias_to_bf16(const float* __restrict__ agg,
                                 const float* __restrict__ b1,
                                 bf16_t* __restrict__ A2, size_t total) {
    size_t i = (size_t)blockIdx.x * blockDim.x + threadIdx.x;
    if (i >= total) return;
    float v = agg[i] + b1[i % HD1];
    v = v > 0.f ? v : (expf(v) - 1.f);
    A2[i] = (bf16_t)v;
}

__global__ void add_bias_inplace(float* __restrict__ out,
                                 const float* __restrict__ b, size_t total, int dim) {
    size_t i = (size_t)blockIdx.x * blockDim.x + threadIdx.x;
    if (i < total) out[i] += b[i % dim];
}

// ---------------------------------------------------------------------------
// Launch
// ---------------------------------------------------------------------------
static inline size_t alignup(size_t x) { return (x + 255) & ~(size_t)255; }

extern "C" void kernel_launch(void* const* d_in, const int* in_sizes, int n_in,
                              void* d_out, int out_size, void* d_ws, size_t ws_size,
                              hipStream_t stream) {
    const float* features = (const float*)d_in[0];
    const int*   esrc     = (const int*)d_in[1];
    const int*   edst     = (const int*)d_in[2];
    const float* W1_src   = (const float*)d_in[3];
    const float* W1_dst   = (const float*)d_in[4];
    const float* a1       = (const float*)d_in[5];
    const float* b1       = (const float*)d_in[6];
    const float* W2_src   = (const float*)d_in[7];
    const float* W2_dst   = (const float*)d_in[8];
    const float* a2       = (const float*)d_in[9];
    const float* b2       = (const float*)d_in[10];
    float* out = (float*)d_out;

    // ---- workspace carve ----
    char* ws = (char*)d_ws;
    size_t cur = 0;
    auto take = [&](size_t bytes) { void* p = ws + cur; cur = alignup(cur + bytes); return p; };

    bf16_t* Xbf    = (bf16_t*)take((size_t)N_NODES * D_IN * 2);
    bf16_t* Wt1    = (bf16_t*)take((size_t)2 * HD1 * D_IN * 2);      // [6144][384]
    bf16_t* Wt2    = (bf16_t*)take((size_t)2 * D2 * HD1 * 2);        // [768][3072]
    float*  F1     = (float*) take((size_t)N_NODES * 2 * HD1 * 4);   // fs|fd layer1
    float*  score1 = (float*) take((size_t)N_EDGES * H1 * 4);
    float*  ex1    = (float*) take((size_t)N_EDGES * H1 * 4);
    float*  smax1  = (float*) take((size_t)N_NODES * H1 * 4);
    float*  den1   = (float*) take((size_t)N_NODES * H1 * 4);
    float*  agg1   = (float*) take((size_t)N_NODES * HD1 * 4);
    bf16_t* A2     = (bf16_t*)take((size_t)N_NODES * HD1 * 2);
    float*  F2     = (float*) take((size_t)N_NODES * 2 * D2 * 4);    // fs|fd layer2
    float*  score2 = (float*) take((size_t)N_EDGES * 4);
    float*  ex2    = (float*) take((size_t)N_EDGES * 4);
    float*  smax2  = (float*) take((size_t)N_NODES * 4);
    float*  den2   = (float*) take((size_t)N_NODES * 4);
    (void)ws_size; (void)in_sizes; (void)n_in; (void)out_size;

    const float NEGINF = -__builtin_huge_valf();
    auto blk = [](size_t n) { return (unsigned)((n + 255) / 256); };

    // ---- prep: bf16 conversions / weight transposes ----
    {
        size_t n = (size_t)N_NODES * D_IN;
        f32_to_bf16<<<blk(n), 256, 0, stream>>>(features, Xbf, n);
    }
    {
        dim3 g((HD1 + 31) / 32, (D_IN + 31) / 32);
        convert_w_transpose<<<g, 256, 0, stream>>>(W1_src, Wt1, D_IN, HD1);
        convert_w_transpose<<<g, 256, 0, stream>>>(W1_dst, Wt1 + (size_t)HD1 * D_IN, D_IN, HD1);
    }
    {
        dim3 g((D2 + 31) / 32, (HD1 + 31) / 32);
        convert_w_transpose<<<g, 256, 0, stream>>>(W2_src, Wt2, HD1, D2);
        convert_w_transpose<<<g, 256, 0, stream>>>(W2_dst, Wt2 + (size_t)D2 * HD1, HD1, D2);
    }

    // ---- layer 1 GEMM: F1[30000][6144] = X[30000][384] x Wt1^T ----
    {
        dim3 g((2 * HD1) / BN, (N_NODES + BM - 1) / BM);
        gemm_bf16_wmma<<<g, 256, 0, stream>>>(Xbf, Wt1, F1, N_NODES, 2 * HD1, D_IN);
    }

    // ---- layer 1 edge softmax + aggregation ----
    fill_f32<<<blk((size_t)N_NODES * H1), 256, 0, stream>>>(smax1, NEGINF, (size_t)N_NODES * H1);
    fill_f32<<<blk((size_t)N_NODES * H1), 256, 0, stream>>>(den1, 0.f, (size_t)N_NODES * H1);
    fill_f32<<<blk((size_t)N_NODES * HD1), 256, 0, stream>>>(agg1, 0.f, (size_t)N_NODES * HD1);

    edge_score_h8<<<N_EDGES, 256, 0, stream>>>(F1, a1, esrc, edst, score1, smax1, N_EDGES);
    edge_exp<H1><<<blk((size_t)N_EDGES * H1), 256, 0, stream>>>(score1, edst, smax1, ex1, den1, N_EDGES);
    edge_agg_h8<<<N_EDGES, 256, 0, stream>>>(F1, ex1, den1, esrc, edst, agg1, N_EDGES);

    // ---- g = elu(agg + b1) -> bf16 ----
    {
        size_t n = (size_t)N_NODES * HD1;
        elu_bias_to_bf16<<<blk(n), 256, 0, stream>>>(agg1, b1, A2, n);
    }

    // ---- layer 2 GEMM: F2[30000][768] = g[30000][3072] x Wt2^T ----
    {
        dim3 g((2 * D2) / BN, (N_NODES + BM - 1) / BM);
        gemm_bf16_wmma<<<g, 256, 0, stream>>>(A2, Wt2, F2, N_NODES, 2 * D2, HD1);
    }

    // ---- layer 2 edge softmax + aggregation (into d_out) ----
    fill_f32<<<blk(N_NODES), 256, 0, stream>>>(smax2, NEGINF, N_NODES);
    fill_f32<<<blk(N_NODES), 256, 0, stream>>>(den2, 0.f, N_NODES);
    fill_f32<<<blk((size_t)N_NODES * D2), 256, 0, stream>>>(out, 0.f, (size_t)N_NODES * D2);

    edge_score_h1<<<(N_EDGES + 7) / 8, 256, 0, stream>>>(F2, a2, esrc, edst, score2, smax2, N_EDGES);
    edge_exp<1><<<blk(N_EDGES), 256, 0, stream>>>(score2, edst, smax2, ex2, den2, N_EDGES);
    edge_agg_h1<<<N_EDGES, 256, 0, stream>>>(F2, ex2, den2, esrc, edst, out, N_EDGES);

    // out = agg2 + b2  (mean over the single head is identity)
    add_bias_inplace<<<blk((size_t)N_NODES * D2), 256, 0, stream>>>(out, b2, (size_t)N_NODES * D2, D2);
}
